// BahdanauAttention_43353399885949
// MI455X (gfx1250) — compile-verified, matrix operand order checked by reading, not run on hardware
//
#include <hip/hip_runtime.h>
#include <hip/hip_bf16.h>

// ---------- problem constants ----------
#define BATCH   64
#define SEQ     4096
#define ENCD    512
#define DECD    512
#define ATTD    128
#define NEGVAL  (-10000000000.0f)

#define SBLK    256          // S rows per block in main kernel
#define KCHUNK  64           // K columns staged in LDS per iteration
#define LDS_STRIDE 72        // padded row stride (bf16 elements)

typedef __attribute__((ext_vector_type(16))) __bf16 v16bf;
typedef __attribute__((ext_vector_type(8)))  float  v8f;

union BF16Frag {
    v16bf v;
    unsigned int u[8];
};

// round-to-nearest-even f32 -> bf16 (bit pattern)
static __device__ __forceinline__ unsigned short f2bf(float f) {
    union { float f; unsigned int u; } p; p.f = f;
    unsigned int r = p.u + 0x7FFFu + ((p.u >> 16) & 1u);
    return (unsigned short)(r >> 16);
}

// Build a 16x32 bf16 fragment from LDS.
// Layout per CDNA5 ISA 7.12.2 (16-bit A 16x32): lane half selects K sub-block,
// VGPR j packs K pair {kOff, kOff+1}.  base points at (row*LDS_STRIDE + kb).
static __device__ __forceinline__ void load_frag(const unsigned short* base,
                                                 BF16Frag& f, int half8) {
#pragma unroll
    for (int j = 0; j < 8; ++j) {
        int ko = (j < 4) ? (half8 + 2 * j) : (16 + half8 + 2 * (j - 4));
        f.u[j] = *(const unsigned int*)(base + ko);
    }
}

// ---------- kernel 1: w1 f32 -> bf16 ([A][K] row-major) ----------
__global__ void cvt_w1_kernel(const float* __restrict__ w1,
                              unsigned short* __restrict__ wbf) {
    int idx = blockIdx.x * blockDim.x + threadIdx.x;   // 128*512 = 65536
    if (idx < ATTD * ENCD) wbf[idx] = f2bf(w1[idx]);
}

// ---------- kernel 2: svec[b][a] = dec[b]·w2[a] + w2_b[a] + w1_b[a] ----------
__global__ void svec_kernel(const float* __restrict__ dec,
                            const float* __restrict__ w2w,
                            const float* __restrict__ w2b,
                            const float* __restrict__ w1b,
                            float* __restrict__ svec) {
    int b = blockIdx.x;
    int a = threadIdx.x;               // 128 threads
    const float* dv = dec + (size_t)b * DECD;
    const float* wv = w2w + (size_t)a * DECD;
    float acc = w2b[a] + w1b[a];
#pragma unroll 8
    for (int k = 0; k < DECD; ++k) acc += dv[k] * wv[k];
    svec[b * ATTD + a] = acc;
}

// ---------- kernel 3: main fused WMMA GEMM + tanh + v-dot + mask ----------
// grid: (SEQ/SBLK, BATCH), block: 256 threads (8 waves)
__global__ void __launch_bounds__(256)
scores_kernel(const float* __restrict__ enc,
              const unsigned short* __restrict__ wbf,  // [ATTD][ENCD] bf16
              const float* __restrict__ svec,          // [B][ATTD]
              const float* __restrict__ vW,            // [ATTD]
              const float* __restrict__ vB,            // [1]
              const int*   __restrict__ mask,          // [B][SEQ]
              float* __restrict__ scores) {            // [B][SEQ]
    __shared__ unsigned short sEnc[SBLK * LDS_STRIDE];   // 36 KB
    __shared__ unsigned short sW[ATTD * LDS_STRIDE];     // 18 KB

    const int b      = blockIdx.y;
    const int sBase  = blockIdx.x * SBLK;
    const int t      = threadIdx.x;
    const int lane   = t & 31;
    const int wave   = t >> 5;
    const int n0     = lane & 15;
    const int half8  = (lane & 16) ? 8 : 0;

    const float* encBase = enc + ((size_t)b * SEQ + sBase) * ENCD;

    v8f acc[2][8];
    {
        v8f z = {};
#pragma unroll
        for (int mt = 0; mt < 2; ++mt)
#pragma unroll
            for (int nt = 0; nt < 8; ++nt) acc[mt][nt] = z;
    }

    const int NKC = ENCD / KCHUNK;   // 8 chunks
    for (int kc = 0; kc < NKC; ++kc) {
        // ---- stage encoder tile: 256 rows x 64 cols, f32 -> bf16 ----
#pragma unroll
        for (int i = 0; i < 16; ++i) {
            int flat = i * 1024 + t * 4;          // 16384 f32 elements
            int row  = flat >> 6;
            int col  = flat & 63;
            float4 v = *(const float4*)(encBase + (size_t)row * ENCD + kc * KCHUNK + col);
            uint2 pk;
            pk.x = (unsigned)f2bf(v.x) | ((unsigned)f2bf(v.y) << 16);
            pk.y = (unsigned)f2bf(v.z) | ((unsigned)f2bf(v.w) << 16);
            *(uint2*)&sEnc[row * LDS_STRIDE + col] = pk;
        }
        // ---- stage w1 tile: 128 rows x 64 cols, bf16 copy ----
#pragma unroll
        for (int i = 0; i < 4; ++i) {
            int flat = i * 2048 + t * 8;          // 8192 bf16 elements
            int a    = flat >> 6;
            int k    = flat & 63;
            uint4 w  = *(const uint4*)(wbf + (size_t)a * ENCD + kc * KCHUNK + k);
            *(uint4*)&sW[a * LDS_STRIDE + k] = w;
        }
        // prefetch next encoder chunk while we compute (global_prefetch_b8)
        if (kc + 1 < NKC)
            __builtin_prefetch(encBase + (size_t)t * ENCD + (kc + 1) * KCHUNK, 0, 1);

        __syncthreads();

        // ---- two K-steps of 32 per chunk ----
#pragma unroll
        for (int ks = 0; ks < 2; ++ks) {
            const int kb = ks * 32;
            BF16Frag a0, a1;
            load_frag(&sEnc[(wave * 32 +  0 + n0) * LDS_STRIDE + kb], a0, half8);
            load_frag(&sEnc[(wave * 32 + 16 + n0) * LDS_STRIDE + kb], a1, half8);
#pragma unroll
            for (int nt = 0; nt < 8; ++nt) {
                BF16Frag bf;
                load_frag(&sW[(nt * 16 + n0) * LDS_STRIDE + kb], bf, half8);
                acc[0][nt] = __builtin_amdgcn_wmma_f32_16x16x32_bf16(
                    false, a0.v, false, bf.v, (short)0, acc[0][nt], false, false);
                acc[1][nt] = __builtin_amdgcn_wmma_f32_16x16x32_bf16(
                    false, a1.v, false, bf.v, (short)0, acc[1][nt], false, false);
            }
        }
        __syncthreads();
    }

    // ---- epilogue: tanh(h + s) · v, lane reduce, mask, store ----
    float svv[8], vwv[8];
#pragma unroll
    for (int nt = 0; nt < 8; ++nt) {
        int N   = nt * 16 + n0;
        svv[nt] = svec[b * ATTD + N];
        vwv[nt] = vW[N];
    }
    const float vb0 = vB[0];

#pragma unroll
    for (int mt = 0; mt < 2; ++mt) {
        int rowBase = sBase + wave * 32 + mt * 16 + ((lane & 16) ? 8 : 0);
#pragma unroll
        for (int r = 0; r < 8; ++r) {
            float p = 0.0f;
#pragma unroll
            for (int nt = 0; nt < 8; ++nt)
                p += tanhf(acc[mt][nt][r] + svv[nt]) * vwv[nt];
            // reduce across the 16-lane N group (wave32)
            p += __shfl_xor(p, 8, 32);
            p += __shfl_xor(p, 4, 32);
            p += __shfl_xor(p, 2, 32);
            p += __shfl_xor(p, 1, 32);
            if (n0 == 0) {
                int row  = rowBase + r;
                float sc = p + vb0;
                int  m   = mask[b * SEQ + row];
                scores[(size_t)b * SEQ + row] = (m == 0) ? NEGVAL : sc;
            }
        }
    }
}

// ---------- kernel 4: softmax over S per batch row -> weights ----------
__global__ void __launch_bounds__(256)
softmax_kernel(const float* __restrict__ scores,
               float* __restrict__ weights) {
    __shared__ float red[256];
    const int b = blockIdx.x;
    const int t = threadIdx.x;
    const float* sc = scores + (size_t)b * SEQ;
    float m = -3.4e38f;
#pragma unroll
    for (int i = 0; i < SEQ / 256; ++i) m = fmaxf(m, sc[t + i * 256]);
    red[t] = m; __syncthreads();
    for (int o = 128; o > 0; o >>= 1) {
        if (t < o) red[t] = fmaxf(red[t], red[t + o]);
        __syncthreads();
    }
    m = red[0]; __syncthreads();
    float s = 0.0f;
#pragma unroll
    for (int i = 0; i < SEQ / 256; ++i) s += __expf(sc[t + i * 256] - m);
    red[t] = s; __syncthreads();
    for (int o = 128; o > 0; o >>= 1) {
        if (t < o) red[t] += red[t + o];
        __syncthreads();
    }
    float inv = 1.0f / red[0];
    float* wp = weights + (size_t)b * SEQ;
#pragma unroll
    for (int i = 0; i < SEQ / 256; ++i) {
        int idx = t + i * 256;
        wp[idx] = __expf(sc[idx] - m) * inv;
    }
}

// ---------- kernel 5: context partials (split S into 8 chunks) ----------
// grid: (8, BATCH), block 256: each thread owns e = t and e = t+256
__global__ void __launch_bounds__(256)
ctx_partial_kernel(const float* __restrict__ enc,
                   const float* __restrict__ weights,
                   float* __restrict__ partial) {      // [B][8][ENCD]
    const int c = blockIdx.x;
    const int b = blockIdx.y;
    const int t = threadIdx.x;
    const int s0 = c * (SEQ / 8);
    const float* encB = enc + ((size_t)b * SEQ + s0) * ENCD;
    const float* wp   = weights + (size_t)b * SEQ + s0;
    float a0 = 0.0f, a1 = 0.0f;
    for (int s = 0; s < SEQ / 8; ++s) {
        float w = wp[s];
        a0 += w * encB[(size_t)s * ENCD + t];
        a1 += w * encB[(size_t)s * ENCD + t + 256];
    }
    float* pp = partial + ((size_t)b * 8 + c) * ENCD;
    pp[t]       = a0;
    pp[t + 256] = a1;
}

// ---------- kernel 6: reduce partials -> context ----------
__global__ void __launch_bounds__(512)
ctx_reduce_kernel(const float* __restrict__ partial,
                  float* __restrict__ ctx) {
    const int b = blockIdx.x;
    const int e = threadIdx.x;     // 512
    float s = 0.0f;
#pragma unroll
    for (int c = 0; c < 8; ++c)
        s += partial[((size_t)b * 8 + c) * ENCD + e];
    ctx[(size_t)b * ENCD + e] = s;
}

extern "C" void kernel_launch(void* const* d_in, const int* in_sizes, int n_in,
                              void* d_out, int out_size, void* d_ws, size_t ws_size,
                              hipStream_t stream) {
    const float* dec   = (const float*)d_in[0];
    const float* enc   = (const float*)d_in[1];
    const int*   amask = (const int*)  d_in[2];
    const float* w1w   = (const float*)d_in[3];
    const float* w1b   = (const float*)d_in[4];
    const float* w2w   = (const float*)d_in[5];
    const float* w2b   = (const float*)d_in[6];
    const float* vw    = (const float*)d_in[7];
    const float* vb    = (const float*)d_in[8];

    float* out = (float*)d_out;
    float* ctx     = out;                        // [B][ENCD]
    float* weights = out + BATCH * ENCD;         // [B][SEQ]

    char* ws = (char*)d_ws;
    unsigned short* wbf  = (unsigned short*)(ws);                       // 64 KB
    float*          svec = (float*)(ws + 65536);                        // 32 KB
    float*          scr  = (float*)(ws + 65536 + 32768);                // 1 MB
    float*          part = (float*)(ws + 65536 + 32768 + 1048576);      // 1 MB

    cvt_w1_kernel<<<128, 512, 0, stream>>>(w1w, wbf);
    svec_kernel<<<BATCH, ATTD, 0, stream>>>(dec, w2w, w2b, w1b, svec);
    scores_kernel<<<dim3(SEQ / SBLK, BATCH), 256, 0, stream>>>(
        enc, wbf, svec, vw, vb, amask, scr);
    softmax_kernel<<<BATCH, 256, 0, stream>>>(scr, weights);
    ctx_partial_kernel<<<dim3(8, BATCH), 256, 0, stream>>>(enc, weights, part);
    ctx_reduce_kernel<<<BATCH, 512, 0, stream>>>(part, ctx);
}
